// GCEncoder_29618094473722
// MI455X (gfx1250) — matrix operand order repaired and checked by reading, not compiled, not run on hardware
//
#include <hip/hip_runtime.h>
#include <hip/hip_bf16.h>

#define N_NODES 4096
#define R_REL   5
#define B_BASIS 30
#define H0      500
#define H0P     512
#define H1      75
#define E_EDGES 262144
#define U_ROWS  2048
#define KDIM    4096
#define EPSV    1e-5f

// GEMM tiling
#define KC         64                 // K-chunk staged in LDS
#define LDS_STRIDE 144                // 128B of K data + 16B pad per column row
#define LDSBUF     (64 * LDS_STRIDE)  // one B-tile buffer (64 cols)

typedef __attribute__((ext_vector_type(16))) __bf16 v16bf;
typedef __attribute__((ext_vector_type(8)))  __bf16 v8bf;
typedef __attribute__((ext_vector_type(8)))  float  v8f;
typedef __attribute__((ext_vector_type(4)))  unsigned int v4u;
typedef __attribute__((ext_vector_type(8)))  int    v8i_;
typedef __attribute__((ext_vector_type(4)))  int    v4i_;

#if defined(__has_builtin)
#  if __has_builtin(__builtin_amdgcn_tensor_load_to_lds)
#    define HAVE_TDM 1
#  endif
#endif
#ifndef HAVE_TDM
#  define HAVE_TDM 0
#endif
#if HAVE_TDM && defined(__has_include)
#  if __has_include(<hip/amd_detail/amd_gfx1250_TDM.h>)
#    define TDM_6ARG 1
#  endif
#endif
#ifndef TDM_6ARG
#  define TDM_6ARG 0
#endif

__device__ __forceinline__ __bf16 f2bf(float f) {
  union { float f; unsigned u; } in; in.f = f;
  unsigned r = (in.u + 0x7fffu + ((in.u >> 16) & 1u)) >> 16;
  union { unsigned short s; __bf16 b; } out; out.s = (unsigned short)r;
  return out.b;
}

#if HAVE_TDM
// Issue a TDM 2D tile load: 64 K-elems (contiguous, bf16) x 64 columns,
// column stride 4096 elems, LDS padded to 144B per column (bank-conflict-free).
__device__ __forceinline__ void tdm_load_b_tile(unsigned int lds_addr,
                                                unsigned long long gaddr) {
  v4u g0;
  g0.x = 1u;                                               // count=1 (valid user D#)
  g0.y = lds_addr;                                         // LDS byte address
  g0.z = (unsigned int)gaddr;                              // global_addr[31:0]
  g0.w = (unsigned int)((gaddr >> 32) & 0x01FFFFFFu) | (2u << 30);  // addr[56:32] | type=2
  v8i_ g1;
  g1[0] = (int)((1u << 16)      // data_size = 2 bytes
              | (1u << 20)      // pad_enable
              | (4u << 22)      // pad_interval: 32 DWORDs (=128B, one column row)
              | (3u << 25));    // pad_amount: 4 DWORDs (=16B)
  g1[1] = (int)(4096u << 16);   // tensor_dim0[15:0]=4096 in bits[31:16]
  g1[2] = (int)(0xFFFFu << 16); // tensor_dim0 hi=0 ; tensor_dim1[15:0]=0xFFFF
  g1[3] = (int)(64u << 16);     // tensor_dim1 hi=0 ; tile_dim0=64
  g1[4] = 64;                   // tile_dim1=64 ; tile_dim2=0
  g1[5] = 4096;                 // tensor_dim0_stride lo (elements)
  g1[6] = 0;                    // stride hi ; tensor_dim1_stride lo
  g1[7] = 0;
  v4i_ z4 = {0, 0, 0, 0};
#if TDM_6ARG
  v8i_ z8 = {0, 0, 0, 0, 0, 0, 0, 0};
  __builtin_amdgcn_tensor_load_to_lds(g0, g1, z4, z4, z8, 0);
#else
  __builtin_amdgcn_tensor_load_to_lds(g0, g1, z4, z4, 0);
#endif
}
#endif

// ---------------- K0: x (f32) -> xb (bf16) ----------------
__global__ void cvt_x_bf16(const float* __restrict__ x, __bf16* __restrict__ xb, int n) {
  int i = blockIdx.x * blockDim.x + threadIdx.x;
  if (i < n) xb[i] = f2bf(x[i]);
}

// ---- K1: WpT[r][o][k] = sum_b comp[r,b]*basis[b,k,o], bf16, o padded to 512 ----
__global__ void build_wpt(const float* __restrict__ basis, const float* __restrict__ comp,
                          __bf16* __restrict__ wpt) {
  __shared__ float sc[R_REL * B_BASIS];
  for (int i = threadIdx.x; i < R_REL * B_BASIS; i += blockDim.x) sc[i] = comp[i];
  __syncthreads();
  int tid = blockIdx.x * blockDim.x + threadIdx.x;   // over N*H0P, o fastest
  int o = tid & (H0P - 1);
  int n = tid >> 9;
  if (n >= N_NODES) return;
  float acc[R_REL] = {};
  if (o < H0) {
    const float* bp = basis + (size_t)n * H0 + o;
    for (int b = 0; b < B_BASIS; ++b) {
      float v = bp[(size_t)b * N_NODES * H0];
      #pragma unroll
      for (int r = 0; r < R_REL; ++r) acc[r] += sc[r * B_BASIS + b] * v;
    }
  }
  #pragma unroll
  for (int r = 0; r < R_REL; ++r)
    wpt[((size_t)r * H0P + o) * KDIM + n] = f2bf(acc[r]);
}

// ---- K1b: rootT[o][k] (bf16, padded) ----
__global__ void build_rootT(const float* __restrict__ root, __bf16* __restrict__ rt) {
  int tid = blockIdx.x * blockDim.x + threadIdx.x;
  int o = tid & (H0P - 1);
  int n = tid >> 9;
  if (n >= N_NODES) return;
  float v = (o < H0) ? root[(size_t)n * H0 + o] : 0.0f;
  rt[(size_t)o * KDIM + n] = f2bf(v);
}

// ---- K2: C = A(4096x4096 bf16 row-major) * B, B given K-major as BT[col][k] ----
// block = 256 threads = 8 waves, tile 128 rows x 64 cols. The 64-col B tile is
// shared by all 8 waves: streamed into LDS by the Tensor Data Mover, double
// buffered (TENSORcnt), WMMA reads B via ds_load_b128, A via global b128.
__global__ __launch_bounds__(256) void gemm_bf16_wmma(
    const __bf16* __restrict__ A, const __bf16* __restrict__ BT, float* __restrict__ C,
    size_t btBatchStride, size_t cBatchStride) {
  __shared__ __align__(16) char sB[2 * LDSBUF];

  const __bf16* bt = BT + (size_t)blockIdx.z * btBatchStride;
  float* c = C + (size_t)blockIdx.z * cBatchStride;
  int waveid = __builtin_amdgcn_readfirstlane((int)(threadIdx.x >> 5));
  int lane = threadIdx.x & 31;
  int lm = lane & 15;       // row (A) / col (B) within fragment
  int kh = lane >> 4;       // half-wave selector per ISA layout
  int mr = blockIdx.x * 128 + (threadIdx.x >> 5) * 16;
  int nc = blockIdx.y * 64;

  // A fragment: elems 0..7 = K[kk + kh*8 .. +8), elems 8..15 = K[kk+16+kh*8 .. +8)
  const __bf16* ap = A + (size_t)(mr + lm) * KDIM + kh * 8;
  unsigned long long gbase =
      (unsigned long long)(uintptr_t)bt + (unsigned long long)nc * KDIM * 2ull;
  unsigned int ldsbase = (unsigned int)(uintptr_t)sB;  // flat->LDS offset (low 32 bits)

  v8f acc[4] = {};
  const int NCH = KDIM / KC;

#if HAVE_TDM
  if (waveid == 0) tdm_load_b_tile(ldsbase, gbase);
#endif

  for (int ch = 0; ch < NCH; ++ch) {
#if HAVE_TDM
    if (waveid == 0) __builtin_amdgcn_s_wait_tensorcnt(0);
    __syncthreads();                       // B tile for chunk ch ready for all waves
    if (waveid == 0 && ch + 1 < NCH)       // prefetch next chunk into other buffer
      tdm_load_b_tile(ldsbase + (unsigned int)(((ch + 1) & 1) * LDSBUF),
                      gbase + (unsigned long long)(ch + 1) * KC * 2ull);
    const char* sBcur = sB + (ch & 1) * LDSBUF;
#else
    __syncthreads();                       // previous compute done, buffer reusable
    {
      int tcol = threadIdx.x >> 2;         // 4 threads per column
      int tsub = (threadIdx.x & 3) * 2;    // each thread copies 2 x 16B
      const v8bf* src = (const v8bf*)(bt + (size_t)(nc + tcol) * KDIM + ch * KC + tsub * 8);
      v8bf* dst = (v8bf*)(sB + (size_t)tcol * LDS_STRIDE + tsub * 16);
      dst[0] = src[0];
      dst[1] = src[1];
    }
    __syncthreads();
    const char* sBcur = sB;
#endif

    const __bf16* apc = ap + ch * KC;
    __builtin_prefetch(apc + 1024, 0, 3);
    #pragma unroll
    for (int half = 0; half < 2; ++half) {
      int kk2 = half * 32;
      v8bf a0 = *(const v8bf*)(apc + kk2);
      v8bf a1 = *(const v8bf*)(apc + kk2 + 16);
      v16bf Af = __builtin_shufflevector(a0, a1,
          0,1,2,3,4,5,6,7,8,9,10,11,12,13,14,15);
      #pragma unroll
      for (int t = 0; t < 4; ++t) {
        // B fragment: col = nc+t*16+lm, elems 0..7 = K[kk2+kh*8 .. +8), elems 8..15 = K[kk2+16+kh*8 .. +8)
        const char* bp = sBcur + (size_t)(t * 16 + lm) * LDS_STRIDE + (kk2 + kh * 8) * 2;
        v8bf b0 = *(const v8bf*)bp;
        v8bf b1 = *(const v8bf*)(bp + 32);
        v16bf Bf = __builtin_shufflevector(b0, b1,
            0,1,2,3,4,5,6,7,8,9,10,11,12,13,14,15);
        acc[t] = __builtin_amdgcn_wmma_f32_16x16x32_bf16(
            false, Af, false, Bf, (short)0, acc[t], false, false);
      }
    }
  }

  // C/D layout: VGPR v -> row mr + kh*8 + v, col nc + t*16 + lm
  #pragma unroll
  for (int t = 0; t < 4; ++t) {
    #pragma unroll
    for (int v = 0; v < 8; ++v) {
      int row = mr + kh * 8 + v;
      int col = nc + t * 16 + lm;
      c[(size_t)row * H0P + col] = acc[t][v];
    }
  }
}

// ---- K3: zero ----
__global__ void zero_f32(float* __restrict__ p, size_t n) {
  size_t i = (size_t)blockIdx.x * blockDim.x + threadIdx.x;
  if (i < n) p[i] = 0.0f;
}

// ---- K4: edge scatter (segment sums + counts) ----
__global__ void edge_scatter(const int* __restrict__ ei, const int* __restrict__ et,
                             const float* __restrict__ h, float* __restrict__ sums,
                             float* __restrict__ cnt) {
  int e = blockIdx.x;
  int s = ei[e];
  int d = ei[E_EDGES + e];
  int t = et[e];
  const float* hp = h + ((size_t)t * N_NODES + s) * H0P;
  float* sp = sums + ((size_t)d * R_REL + t) * H0P;
  for (int o = threadIdx.x; o < H0; o += blockDim.x)
    atomicAdd(&sp[o], hp[o]);
  if (threadIdx.x == 0) atomicAdd(&cnt[d * R_REL + t], 1.0f);
}

// ---- K5: feats = segment means summed over r + x@root (already in feats) + bias ----
__global__ void combine(const float* __restrict__ sums, const float* __restrict__ cnt,
                        const float* __restrict__ bias, float* __restrict__ feats) {
  int tid = blockIdx.x * blockDim.x + threadIdx.x;
  int o = tid & (H0P - 1);
  int n = tid >> 9;
  if (n >= N_NODES || o >= H0) return;
  float acc = 0.0f;
  #pragma unroll
  for (int r = 0; r < R_REL; ++r) {
    int seg = n * R_REL + r;
    acc += sums[(size_t)seg * H0P + o] / fmaxf(cnt[seg], 1.0f);
  }
  feats[(size_t)n * H0P + o] += acc + bias[o];
}

// ---- K6: z = feats @ fc_w^T, then per-row BN + ReLU, fused. Block per node. ----
__global__ __launch_bounds__(128) void fc_bn_relu(
    const float* __restrict__ feats, const float* __restrict__ fc_w,
    const float* __restrict__ gu, const float* __restrict__ bu,
    const float* __restrict__ gi, const float* __restrict__ bi,
    float* __restrict__ out) {
  __shared__ float srow[H0];
  __shared__ float red[128];
  int n = blockIdx.x;
  const float* fr = feats + (size_t)n * H0P;
  for (int o = threadIdx.x; o < H0; o += blockDim.x) srow[o] = fr[o];
  __syncthreads();
  int j = threadIdx.x;
  float z = 0.0f;
  if (j < H1) {
    const float* w = fc_w + (size_t)j * H0;
    for (int o = 0; o < H0; ++o) z += srow[o] * w[o];
  }
  red[threadIdx.x] = (j < H1) ? z : 0.0f;
  __syncthreads();
  for (int s = 64; s > 0; s >>= 1) {
    if (threadIdx.x < s) red[threadIdx.x] += red[threadIdx.x + s];
    __syncthreads();
  }
  float mu = red[0] / (float)H1;
  __syncthreads();
  float dz = (j < H1) ? (z - mu) : 0.0f;
  red[threadIdx.x] = dz * dz;
  __syncthreads();
  for (int s = 64; s > 0; s >>= 1) {
    if (threadIdx.x < s) red[threadIdx.x] += red[threadIdx.x + s];
    __syncthreads();
  }
  float var = red[0] / (float)H1;
  if (j < H1) {
    float g, b;
    if (n < U_ROWS) { g = gu[n]; b = bu[n]; }
    else            { g = gi[n - U_ROWS]; b = bi[n - U_ROWS]; }
    float y = g * (z - mu) * rsqrtf(var + EPSV) + b;
    out[(size_t)n * H1 + j] = fmaxf(y, 0.0f);
  }
}

extern "C" void kernel_launch(void* const* d_in, const int* in_sizes, int n_in,
                              void* d_out, int out_size, void* d_ws, size_t ws_size,
                              hipStream_t stream) {
  (void)in_sizes; (void)n_in; (void)out_size;
  const float* x     = (const float*)d_in[0];
  const float* basis = (const float*)d_in[1];
  const float* comp  = (const float*)d_in[2];
  const float* root  = (const float*)d_in[3];
  const float* bias  = (const float*)d_in[4];
  const float* fc_w  = (const float*)d_in[5];
  const float* gu    = (const float*)d_in[6];
  const float* bu    = (const float*)d_in[7];
  const float* gi    = (const float*)d_in[8];
  const float* bi    = (const float*)d_in[9];
  const int*   ei    = (const int*)d_in[10];
  const int*   et    = (const int*)d_in[11];
  float* out = (float*)d_out;

  char* ws = (char*)d_ws;
  size_t off = 0;
  auto take = [&](size_t bytes) -> char* {
    char* p = ws + off;
    off = (off + bytes + 255) & ~(size_t)255;
    return p;
  };
  __bf16* xb    = (__bf16*)take((size_t)N_NODES * KDIM * 2);
  __bf16* wpt   = (__bf16*)take((size_t)R_REL * H0P * KDIM * 2);
  __bf16* rt    = (__bf16*)take((size_t)H0P * KDIM * 2);
  float*  h     = (float*)take((size_t)R_REL * N_NODES * H0P * 4);
  float*  sums  = (float*)take((size_t)N_NODES * R_REL * H0P * 4);
  float*  cnt   = (float*)take((size_t)N_NODES * R_REL * 4);
  float*  feats = (float*)take((size_t)N_NODES * H0P * 4);
  if (off > ws_size) return;  // workspace too small; nothing safe to do

  // 1) casts / weight combine
  int nx = N_NODES * KDIM;
  cvt_x_bf16<<<nx / 256, 256, 0, stream>>>(x, xb, nx);
  build_wpt<<<(N_NODES * H0P) / 256, 256, 0, stream>>>(basis, comp, wpt);
  build_rootT<<<(N_NODES * H0P) / 256, 256, 0, stream>>>(root, rt);

  // 2) GEMMs: h[r] = x @ W[r]  (batched over r), feats = x @ root
  dim3 gH(N_NODES / 128, H0P / 64, R_REL);
  gemm_bf16_wmma<<<gH, 256, 0, stream>>>(xb, wpt, h,
                                         (size_t)H0P * KDIM, (size_t)N_NODES * H0P);
  dim3 gR(N_NODES / 128, H0P / 64, 1);
  gemm_bf16_wmma<<<gR, 256, 0, stream>>>(xb, rt, feats, 0, 0);

  // 3) segment mean over edges
  size_t nsum = (size_t)N_NODES * R_REL * H0P;
  zero_f32<<<(unsigned)((nsum + 255) / 256), 256, 0, stream>>>(sums, nsum);
  size_t ncnt = (size_t)N_NODES * R_REL;
  zero_f32<<<(unsigned)((ncnt + 255) / 256), 256, 0, stream>>>(cnt, ncnt);
  edge_scatter<<<E_EDGES, 128, 0, stream>>>(ei, et, h, sums, cnt);
  combine<<<(N_NODES * H0P) / 256, 256, 0, stream>>>(sums, cnt, bias, feats);

  // 4) FC + BN + ReLU fused; u rows [0,2048), i rows [2048,4096) -> row-major d_out
  fc_bn_relu<<<N_NODES, 128, 0, stream>>>(feats, fc_w, gu, bu, gi, bi, out);
}
